// GenericTFB_5703716569215
// MI455X (gfx1250) — matrix operand order repaired and checked by reading, not compiled, not run on hardware
//
#include <hip/hip_runtime.h>
#include <hip/hip_bf16.h>

// ---------------------------------------------------------------------------
// Transformer block (no-softmax attention) for MI455X / gfx1250, bf16 WMMA.
// attention rewritten as Q @ (K^T V): K^T V is 32x32 per head.
// All WMMA B-matrices pre-swizzled to fragment-ready layout (32B/lane loads).
// MLP activation staging uses CDNA5 async global->LDS transfers (ASYNCcnt).
// ---------------------------------------------------------------------------

typedef __attribute__((ext_vector_type(16))) __bf16 v16bf;
typedef __attribute__((ext_vector_type(8)))  float  v8f;

#define JV    1025   // valid sequence length
#define JPAD  1056   // padded to multiple of 32 (66 tiles of 16)
#define MDIM  256
#define NBI   16     // B*I
#define HEADS 8

union FragBF { v16bf v; uint4 q[2]; unsigned short u[16]; };

static __device__ __forceinline__ unsigned short f32_to_bf16(float f) {
    unsigned int u = __float_as_uint(f);
    u = (u + 0x7FFFu + ((u >> 16) & 1u)) >> 16;
    return (unsigned short)u;
}

static __device__ __forceinline__ v8f wmma_bf16(v16bf a, v16bf b, v8f c) {
    return __builtin_amdgcn_wmma_f32_16x16x32_bf16(
        /*neg_a=*/false, a, /*neg_b=*/false, b,
        /*c_mod=*/(short)0, c, /*reuse_a=*/false, /*reuse_b=*/false);
}

// A fragment (16x32 bf16, M x K) from a row-major source, leading dim ld.
// Layout: lanes 0-15 -> M=lane, K = 0..15 (idx0-15);
//         lanes 16-31 -> M=lane-16, K = 16..31.
// Two contiguous 16-byte runs per lane -> 2x b128 loads.
static __device__ __forceinline__ v16bf load_a_frag(
    const unsigned short* __restrict__ base, int ld, int row0, int k0, int lane) {
    const int row = row0 + (lane & 15);
    const int hi  = (lane >> 4) * 16;
    const unsigned short* p = base + (long)row * ld + k0 + hi;
    FragBF f;
    f.q[0] = *(const uint4*)(p);
    f.q[1] = *(const uint4*)(p + 8);
    return f.v;
}

// Fragment-ready load: 32 contiguous bytes per lane.
static __device__ __forceinline__ v16bf load_frag16(
    const unsigned short* __restrict__ p0, int lane) {
    const uint4* q = (const uint4*)(p0 + (lane << 4));
    FragBF f;
    f.q[0] = q[0];
    f.q[1] = q[1];
    return f.v;
}

// Packed-B tile load: base holds tiles of 512 ushorts (32 lanes x 16 bf16).
static __device__ __forceinline__ v16bf load_b_packed(
    const unsigned short* __restrict__ base, long tile, int lane) {
    return load_frag16(base + tile * 512, lane);
}

static __device__ __forceinline__ v8f zero8() {
    v8f z = {0.f, 0.f, 0.f, 0.f, 0.f, 0.f, 0.f, 0.f};
    return z;
}

// Async global->LDS b128 copy (CDNA5). dst_lds = LDS byte offset, src = global.
// Tracked by ASYNCcnt; pair with s_wait_asynccnt.
static __device__ __forceinline__ void async_copy_b128(unsigned dst_lds,
                                                       const void* src) {
    asm volatile("global_load_async_to_lds_b128 %0, %1, off"
                 :: "v"(dst_lds), "v"((unsigned long long)src)
                 : "memory");
}
static __device__ __forceinline__ void async_wait0() {
    asm volatile("s_wait_asynccnt 0" ::: "memory");
}

// ---------------------------------------------------------------------------
// Kernel 1: weight prep. Pack B matrices into WMMA fragment-ready tiles:
// dst[((kt*NT + nt)*32 + lane)*16 + i], lane = (n&15)|(k&16), i = k&15.
// ---------------------------------------------------------------------------
__global__ void tfb_prep(const float* __restrict__ Wqkv, const float* __restrict__ Wo,
                         const float* __restrict__ fc1_w, const float* __restrict__ fc2_w,
                         unsigned short* __restrict__ Bqkv,   // packed [256 x 768], NT=48
                         unsigned short* __restrict__ B1,     // packed [256 x 1024], NT=64
                         unsigned short* __restrict__ B2,     // packed [1024 x 256], NT=16
                         float* __restrict__ wo_row) {        // [256]
    const long tid = (long)blockIdx.x * blockDim.x + threadIdx.x;
    const long stride = (long)gridDim.x * blockDim.x;
    // Bqkv: K=256 (m), N=768 (q*256+h*32+d); src Wqkv[q][hd][m]
    for (long o = tid; o < 256L * 768; o += stride) {
        const int i = (int)(o & 15), lane = (int)((o >> 4) & 31), t = (int)(o >> 9);
        const int kt = t / 48, nt = t % 48;
        const int k = kt * 32 + ((lane >> 4) << 4) + i;
        const int n = nt * 16 + (lane & 15);
        Bqkv[o] = f32_to_bf16(Wqkv[(long)n * 256 + k]);  // (q*256+hd)*256 + m
    }
    // B1: K=256, N=1024; src fc1_w[n][k]
    for (long o = tid; o < 256L * 1024; o += stride) {
        const int i = (int)(o & 15), lane = (int)((o >> 4) & 31), t = (int)(o >> 9);
        const int kt = t >> 6, nt = t & 63;
        const int k = kt * 32 + ((lane >> 4) << 4) + i;
        const int n = nt * 16 + (lane & 15);
        B1[o] = f32_to_bf16(fc1_w[(long)n * 256 + k]);
    }
    // B2: K=1024, N=256; src fc2_w[n][k]
    for (long o = tid; o < 1024L * 256; o += stride) {
        const int i = (int)(o & 15), lane = (int)((o >> 4) & 31), t = (int)(o >> 9);
        const int kt = t >> 4, nt = t & 15;
        const int k = kt * 32 + ((lane >> 4) << 4) + i;
        const int n = nt * 16 + (lane & 15);
        B2[o] = f32_to_bf16(fc2_w[(long)n * 1024 + k]);
    }
    for (long m = tid; m < 256; m += stride) {
        float s = 0.f;
        for (int n = 0; n < 256; ++n) s += Wo[m * 256 + n];
        wo_row[m] = s;
    }
}

// ---------------------------------------------------------------------------
// Kernel 2/6: LayerNorm over last dim (256), fp32 in -> bf16 padded out.
// ---------------------------------------------------------------------------
__global__ void tfb_layernorm(const float* __restrict__ in, const float* __restrict__ g,
                              const float* __restrict__ b, unsigned short* __restrict__ out) {
    const int wid = threadIdx.x >> 5, lane = threadIdx.x & 31;
    const long row = (long)blockIdx.x * 8 + wid;
    const int bi = (int)(row / JPAD), j = (int)(row % JPAD);
    unsigned short* op = out + ((long)bi * JPAD + j) * MDIM + lane * 8;
    if (j >= JV) {
#pragma unroll
        for (int i = 0; i < 8; ++i) op[i] = 0;
        return;
    }
    const float* ip = in + ((long)bi * JV + j) * MDIM + lane * 8;
    float x[8];
    float s = 0.f;
#pragma unroll
    for (int i = 0; i < 8; ++i) { x[i] = ip[i]; s += x[i]; }
#pragma unroll
    for (int off = 16; off >= 1; off >>= 1) s += __shfl_xor(s, off, 32);
    const float mu = s * (1.f / 256.f);
    float vs = 0.f;
#pragma unroll
    for (int i = 0; i < 8; ++i) { const float d = x[i] - mu; vs += d * d; }
#pragma unroll
    for (int off = 16; off >= 1; off >>= 1) vs += __shfl_xor(vs, off, 32);
    const float rinv = rsqrtf(vs * (1.f / 256.f) + 1e-5f);
#pragma unroll
    for (int i = 0; i < 8; ++i) {
        const int c = lane * 8 + i;
        op[i] = f32_to_bf16((x[i] - mu) * rinv * g[c] + b[c]);
    }
}

// ---------------------------------------------------------------------------
// Kernel 3: QKV GEMM per (b,i): [JPADx256] x [256x768] -> qkv bf16 [JPADx768]
// block = 128 (4 waves); each wave owns a 16x32 output tile (A-frag reuse).
// ---------------------------------------------------------------------------
__global__ void tfb_qkv_gemm(const unsigned short* __restrict__ sln,
                             const unsigned short* __restrict__ Bqkv,
                             unsigned short* __restrict__ qkv) {
    const int w = threadIdx.x >> 5, lane = threadIdx.x & 31;
    const int mtile = blockIdx.x;                 // 0..65
    const int pair  = blockIdx.y * 4 + w;         // 0..23
    const int bi = blockIdx.z;
    const unsigned short* A = sln + (long)bi * JPAD * MDIM;
    v8f acc[2] = {zero8(), zero8()};
#pragma unroll
    for (int kc = 0; kc < 8; ++kc) {
        if (kc < 7)
            __builtin_prefetch(Bqkv + (((long)(kc + 1) * 48 + 2 * pair) * 32 + lane) * 16, 0, 0);
        const v16bf a = load_a_frag(A, MDIM, mtile * 16, kc * 32, lane);
#pragma unroll
        for (int et = 0; et < 2; ++et) {
            const v16bf b = load_b_packed(Bqkv, (long)kc * 48 + 2 * pair + et, lane);
            acc[et] = wmma_bf16(a, b, acc[et]);
        }
    }
    const int M0 = (lane >> 4) * 8;
    unsigned short* O = qkv + (long)bi * JPAD * 768;
#pragma unroll
    for (int et = 0; et < 2; ++et) {
        const int col = (2 * pair + et) * 16 + (lane & 15);
#pragma unroll
        for (int r = 0; r < 8; ++r)
            O[(long)(mtile * 16 + M0 + r) * 768 + col] = f32_to_bf16(acc[et][r]);
    }
}

// ---------------------------------------------------------------------------
// Kernel 4: S = scale * K^T V per (b,i,h). K/V scattered into fragment-ready
// LDS layouts; result stored in packed-B layout for the imv kernel.
// ---------------------------------------------------------------------------
__global__ void tfb_ktv(const unsigned short* __restrict__ qkv,
                        unsigned short* __restrict__ S, float scale) {
    __shared__ __align__(16) unsigned short ktf[2 * 512];  // A frags, dtile 0/1
    __shared__ __align__(16) unsigned short vtf[2 * 512];  // B frags, etile 0/1
    const int lane = threadIdx.x;                          // block = 32 (one wave)
    const int bi = blockIdx.x >> 3, h = blockIdx.x & 7;
    const unsigned short* base = qkv + (long)bi * JPAD * 768;
    v8f acc[2][2];
    for (int dt = 0; dt < 2; ++dt)
        for (int et = 0; et < 2; ++et) acc[dt][et] = zero8();

    // this lane supplies K-index j = jc+lane; its A-layout position:
    const int aidx = lane & 15;   // idx within lane's 16
    const int ahib = lane & 16;            // hi-lane bit
    for (int jc = 0; jc < JPAD; jc += 32) {
        const long j = jc + lane;
        const unsigned short* krow = base + j * 768 + 256 + h * 32;
        const unsigned short* vrow = base + j * 768 + 512 + h * 32;
#pragma unroll
        for (int d = 0; d < 32; ++d)   // A element (M=d, K=j)
            ktf[((d >> 4) << 9) + ((((d & 15) | ahib)) << 4) + aidx] = krow[d];
#pragma unroll
        for (int e = 0; e < 32; ++e)   // B element (K=j, N=e)
            vtf[((e >> 4) << 9) + (((e & 15) | (lane & 16)) << 4) + (lane & 15)] = vrow[e];
        __syncthreads();
#pragma unroll
        for (int dt = 0; dt < 2; ++dt) {
            const v16bf a = load_frag16(ktf + (dt << 9), lane);
#pragma unroll
            for (int et = 0; et < 2; ++et) {
                const v16bf b = load_frag16(vtf + (et << 9), lane);
                acc[dt][et] = wmma_bf16(a, b, acc[dt][et]);
            }
        }
        __syncthreads();
    }
    // store in packed-B layout (K dim = d (1 ktile of 32), N dim = e (2 ntiles)):
    // dst lane_id = n | (dt<<4), idx = d&15 = M0+r
    unsigned short* Sp = S + (long)blockIdx.x * 1024;
    const int M0 = (lane >> 4) * 8, n = lane & 15;
#pragma unroll
    for (int dt = 0; dt < 2; ++dt)
#pragma unroll
        for (int et = 0; et < 2; ++et)
#pragma unroll
            for (int r = 0; r < 8; ++r)
                Sp[(et * 32 + (n | (dt << 4))) * 16 + M0 + r] =
                    f32_to_bf16(acc[dt][et][r] * scale);
}

// ---------------------------------------------------------------------------
// Kernel 5: imv = Q @ S ; s2 = wo_row * imv + savespace (fp32 out)
// ---------------------------------------------------------------------------
__global__ void tfb_imv_s2(const unsigned short* __restrict__ qkv,
                           const unsigned short* __restrict__ S,
                           const float* __restrict__ wo_row,
                           const float* __restrict__ savespace,
                           float* __restrict__ s2) {
    const int h = threadIdx.x >> 5, lane = threadIdx.x & 31;
    const int jtile = blockIdx.x, bi = blockIdx.y;
    const unsigned short* Q = qkv + (long)bi * JPAD * 768;  // q = cols 0..255
    const unsigned short* Sp = S + (long)(bi * 8 + h) * 1024;
    const v16bf a = load_a_frag(Q, 768, jtile * 16, h * 32, lane);
    v8f acc[2];
#pragma unroll
    for (int et = 0; et < 2; ++et) {
        const v16bf b = load_b_packed(Sp, et, lane);
        acc[et] = wmma_bf16(a, b, zero8());
    }
    const int M0 = (lane >> 4) * 8, n = lane & 15;
#pragma unroll
    for (int et = 0; et < 2; ++et) {
        const int col = h * 32 + et * 16 + n;
        const float wr = wo_row[col];
#pragma unroll
        for (int r = 0; r < 8; ++r) {
            const int j = jtile * 16 + M0 + r;
            if (j < JV) {
                const long idx = ((long)bi * JV + j) * MDIM + col;
                s2[idx] = wr * acc[et][r] + savespace[idx];
            }
        }
    }
}

// ---------------------------------------------------------------------------
// Kernel 7: fused MLP + residual. Hidden (1024) in 4 chunks of 256 via LDS.
// Input rows staged with async global->LDS (ASYNCcnt) instead of VGPR bounce.
// ---------------------------------------------------------------------------
__global__ void tfb_mlp(const unsigned short* __restrict__ hln,
                        const unsigned short* __restrict__ B1,
                        const float* __restrict__ fc1_b,
                        const unsigned short* __restrict__ B2,
                        const float* __restrict__ fc2_b,
                        const float* __restrict__ s2,
                        float* __restrict__ out) {
    __shared__ __align__(16) unsigned short aln[16 * 256];
    __shared__ __align__(16) unsigned short ht[16 * 256];
    const int w = threadIdx.x >> 5, lane = threadIdx.x & 31;
    const int jtile = blockIdx.x, bi = blockIdx.y;
    const int M0 = (lane >> 4) * 8, n = lane & 15;

    {   // stage 16x256 bf16 input rows (8 KB) via async global->LDS b128
        const unsigned short* gsrc = hln + ((long)bi * JPAD + jtile * 16) * MDIM;
        const unsigned lds0 = (unsigned)(unsigned long long)(&aln[0]);  // LDS = addr[31:0]
        for (int t = threadIdx.x; t < 512; t += 256)
            async_copy_b128(lds0 + t * 16, gsrc + t * 8);
        async_wait0();
    }
    __syncthreads();

    v8f oacc[2] = {zero8(), zero8()};
    for (int t = 0; t < 4; ++t) {
        const int hbase = t * 256;
        const int hc0 = hbase + w * 32;
        v8f hacc[2] = {zero8(), zero8()};
#pragma unroll
        for (int kc = 0; kc < 8; ++kc) {
            const v16bf a = load_a_frag(aln, 256, 0, kc * 32, lane);
#pragma unroll
            for (int et = 0; et < 2; ++et) {
                // B1 packed: tile = kt*64 + nt, nt = hc0/16 + et
                const v16bf b = load_b_packed(B1, (long)kc * 64 + (hc0 >> 4) + et, lane);
                hacc[et] = wmma_bf16(a, b, hacc[et]);
            }
        }
#pragma unroll
        for (int et = 0; et < 2; ++et) {
            const int colh = hc0 + et * 16 + n;
            const float bias = fc1_b[colh];
            const int lcol = w * 32 + et * 16 + n;
#pragma unroll
            for (int r = 0; r < 8; ++r) {
                const float x = hacc[et][r] + bias;
                const float ge = 0.5f * x * (1.0f + erff(x * 0.70710678118654752f));
                ht[(M0 + r) * 256 + lcol] = f32_to_bf16(ge);
            }
        }
        __syncthreads();
#pragma unroll
        for (int kc = 0; kc < 8; ++kc) {
            const int ktg = t * 8 + kc;      // global K tile in 0..31
            if (kc < 7)
                __builtin_prefetch(B2 + (((long)(ktg + 1) * 16 + w * 2) * 32 + lane) * 16, 0, 0);
            const v16bf a = load_a_frag(ht, 256, 0, kc * 32, lane);
#pragma unroll
            for (int et = 0; et < 2; ++et) {
                const v16bf b = load_b_packed(B2, (long)ktg * 16 + w * 2 + et, lane);
                oacc[et] = wmma_bf16(a, b, oacc[et]);
            }
        }
        __syncthreads();
    }
#pragma unroll
    for (int et = 0; et < 2; ++et) {
        const int col = w * 32 + et * 16 + n;
        const float bias = fc2_b[col];
#pragma unroll
        for (int r = 0; r < 8; ++r) {
            const int j = jtile * 16 + M0 + r;
            if (j < JV) {
                const long idx = ((long)bi * JV + j) * MDIM + col;
                out[idx] = oacc[et][r] + bias + s2[idx];
            }
        }
    }
}

// ---------------------------------------------------------------------------
// Host launcher
// ---------------------------------------------------------------------------
extern "C" void kernel_launch(void* const* d_in, const int* in_sizes, int n_in,
                              void* d_out, int out_size, void* d_ws, size_t ws_size,
                              hipStream_t stream) {
    const float* savespace = (const float*)d_in[1];
    const float* Wqkv      = (const float*)d_in[2];
    const float* Wo        = (const float*)d_in[3];
    const float* ln1_g     = (const float*)d_in[4];
    const float* ln1_b     = (const float*)d_in[5];
    const float* ln2_g     = (const float*)d_in[6];
    const float* ln2_b     = (const float*)d_in[7];
    const float* fc1_w     = (const float*)d_in[8];
    const float* fc1_b     = (const float*)d_in[9];
    const float* fc2_w     = (const float*)d_in[10];
    const float* fc2_b     = (const float*)d_in[11];
    float* out = (float*)d_out;

    char* ws = (char*)d_ws;
    size_t off = 0;
    auto carve = [&](size_t bytes) -> void* {
        off = (off + 255) & ~(size_t)255;
        void* p = ws + off;
        off += bytes;
        return p;
    };
    unsigned short* sln  = (unsigned short*)carve((size_t)NBI * JPAD * MDIM * 2);
    unsigned short* qkv  = (unsigned short*)carve((size_t)NBI * JPAD * 768 * 2);
    unsigned short* hln  = (unsigned short*)carve((size_t)NBI * JPAD * MDIM * 2);
    unsigned short* Smat = (unsigned short*)carve((size_t)NBI * HEADS * 32 * 32 * 2);
    float*          s2   = (float*)carve((size_t)NBI * JV * MDIM * 4);
    unsigned short* Bqkv = (unsigned short*)carve((size_t)256 * 768 * 2);
    unsigned short* B1w  = (unsigned short*)carve((size_t)256 * 1024 * 2);
    unsigned short* B2w  = (unsigned short*)carve((size_t)1024 * 256 * 2);
    float*          worw = (float*)carve((size_t)256 * 4);

    const float scale = 0.17677669529663687f;  // 1/sqrt(32)

    tfb_prep<<<512, 256, 0, stream>>>(Wqkv, Wo, fc1_w, fc2_w, Bqkv, B1w, B2w, worw);
    tfb_layernorm<<<(NBI * JPAD) / 8, 256, 0, stream>>>(savespace, ln1_g, ln1_b, sln);
    tfb_qkv_gemm<<<dim3(JPAD / 16, 6, NBI), 128, 0, stream>>>(sln, Bqkv, qkv);
    tfb_ktv<<<NBI * HEADS, 32, 0, stream>>>(qkv, Smat, scale);
    tfb_imv_s2<<<dim3(JPAD / 16, NBI), 256, 0, stream>>>(qkv, Smat, worw, savespace, s2);
    tfb_layernorm<<<(NBI * JPAD) / 8, 256, 0, stream>>>(s2, ln2_g, ln2_b, hln);
    tfb_mlp<<<dim3(JPAD / 16, NBI), 256, 0, stream>>>(hln, B1w, fc1_b, B2w, fc2_b, s2, out);
}
